// MultiHeadAttention_43078521979363
// MI455X (gfx1250) — matrix operand order repaired and checked by reading, not compiled, not run on hardware
//
#include <hip/hip_runtime.h>
#include <hip/hip_bf16.h>

typedef __attribute__((ext_vector_type(16))) _Float16 v16h;
typedef __attribute__((ext_vector_type(8)))  float    v8f;
typedef __attribute__((ext_vector_type(4)))  float    f32x4;
typedef __attribute__((ext_vector_type(4)))  unsigned u32x4;
typedef __attribute__((ext_vector_type(2)))  unsigned u32x2;
typedef __attribute__((ext_vector_type(4)))  unsigned tdm_u4;
typedef __attribute__((ext_vector_type(8)))  int      tdm_i8;
typedef __attribute__((ext_vector_type(4)))  int      tdm_i4;

#if defined(__has_builtin)
#if __has_builtin(__builtin_amdgcn_tensor_load_to_lds) && \
    __has_builtin(__builtin_amdgcn_s_wait_tensorcnt)
#define USE_TDM 1
#endif
#endif
#ifndef USE_TDM
#define USE_TDM 0
#endif

union Frag16 { v16h h; u32x4 q[2]; };

__device__ inline v8f wmma16x16x32(v16h a, v16h b, v8f c) {
  // D = A(16x32 f16) * B(32x16 f16) + C(16x16 f32)
  return __builtin_amdgcn_wmma_f32_16x16x32_f16(false, a, false, b, (short)0, c,
                                                false, false);
}

// XOR butterfly within 16-lane half-groups via ds_swizzle (imm-encoded mask).
template <int IMM>
__device__ inline float swz_xor(float x) {
  return __int_as_float(__builtin_amdgcn_ds_swizzle(__float_as_int(x), IMM));
}
__device__ inline float red_max16(float x) {
  x = fmaxf(x, swz_xor<0x041f>(x));   // xor 1
  x = fmaxf(x, swz_xor<0x081f>(x));   // xor 2
  x = fmaxf(x, swz_xor<0x101f>(x));   // xor 4
  x = fmaxf(x, swz_xor<0x201f>(x));   // xor 8
  return x;
}
__device__ inline float red_sum16(float x) {
  x += swz_xor<0x041f>(x);
  x += swz_xor<0x081f>(x);
  x += swz_xor<0x101f>(x);
  x += swz_xor<0x201f>(x);
  return x;
}

#if USE_TDM
__device__ inline unsigned lds_off_of(const void* p) {
  return (unsigned)(unsigned long long)(uintptr_t)p;  // low 32 bits = LDS byte offset
}
// TDM: async copy of a [tile_h x tile_w] f16 tile (row stride 'stride' f16)
// from global into LDS at lds_off, padding each 64-f16 row to 72 f16
// (pad_interval=4 -> every 32 DWORDs, pad_amount=3 -> +4 DWORDs).
__device__ inline void tdm_load_tile_f16(unsigned lds_off, const void* gsrc,
                                         int tile_w, int tile_h, int stride) {
  unsigned long long ga = (unsigned long long)(uintptr_t)gsrc;
  tdm_u4 g0;
  g0[0] = 1u;                                   // count=1 (user descriptor)
  g0[1] = lds_off;                              // lds_addr (bytes)
  g0[2] = (unsigned)ga;                         // global_addr[31:0]
  g0[3] = (unsigned)(ga >> 32) | (2u << 30);    // global_addr[56:32] | type=2
  tdm_i8 g1;
  g1[0] = (1 << 16)                             // data_size = 2 bytes
        | (1 << 20)                             // pad_enable
        | (4 << 22)                             // pad_interval: 32 DWORDs
        | (3 << 25);                            // pad_amount: 4 DWORDs (8 f16)
  g1[1] = (tile_w & 0xffff) << 16;              // tensor_dim0[15:0] @ [63:48]
  g1[2] = (tile_h & 0xffff) << 16;              // tensor_dim1[15:0] @ [95:80]
  g1[3] = (tile_w & 0xffff) << 16;              // tile_dim0 @ [127:112]
  g1[4] = (tile_h & 0xffff);                    // tile_dim1 @ [143:128]
  g1[5] = stride;                               // tensor_dim0_stride[31:0]
  g1[6] = 0;
  g1[7] = 0;
  tdm_i4 z4 = {0, 0, 0, 0};
#if __clang_major__ >= 23
  tdm_i8 z8 = {};
  __builtin_amdgcn_tensor_load_to_lds(g0, g1, z4, z4, z8, 0);
#else
  __builtin_amdgcn_tensor_load_to_lds(g0, g1, z4, z4, 0);
#endif
}
#endif  // USE_TDM

// ---------------------------------------------------------------------------
// GEMM: Y[M x N] = X[M x K] @ W[K x N] + bias   (W, bias always f32)
// 256 threads (8 waves), tile 128(M) x 64(N), K-step 32; waves 4(M) x 2(N),
// each wave 32x32 via 2x2 WMMA frags. LDS rows padded to 40 f16 (80B):
// 16B-aligned ds_load_b128 gathers, conflict-free across 16 rows.
// ---------------------------------------------------------------------------
template <typename XT, typename YT>
__global__ __launch_bounds__(256)
void gemm_bias_kernel(const XT* __restrict__ X, const float* __restrict__ W,
                      const float* __restrict__ bias, YT* __restrict__ Y,
                      int M, int N, int K) {
  constexpr int XS = 40;
  __shared__ _Float16 Xs[128 * XS];            // A tile, row-major [m][k]
  __shared__ _Float16 Wn[64 * XS];             // B tile, n-major  [n][k]

  const int tid  = threadIdx.x;
  const int lane = tid & 31, wave = tid >> 5;
  const int lm   = lane & 15, half = lane >> 4;
  const int wm   = wave & 3,  wn   = wave >> 2;
  const int mbase = blockIdx.y * 128;
  const int nbase = blockIdx.x * 64;

  v8f acc[2][2] = {};

  for (int k0 = 0; k0 < K; k0 += 32) {
#pragma unroll
    for (int i = 0; i < 4; ++i) {
      int linear = tid + i * 256;              // float4 units, 0..1023
      int row = linear >> 3;
      int c4  = linear & 7;
      if constexpr (sizeof(XT) == 4) {
        f32x4 xv = *(const f32x4*)&X[(mbase + row) * K + k0 + c4 * 4];
        _Float16* dst = &Xs[row * XS + c4 * 4];
        dst[0] = (_Float16)xv[0]; dst[1] = (_Float16)xv[1];
        dst[2] = (_Float16)xv[2]; dst[3] = (_Float16)xv[3];
      } else {
        *(u32x2*)&Xs[row * XS + c4 * 4] =
            *(const u32x2*)&X[(mbase + row) * K + k0 + c4 * 4];
      }
      if (k0 + 32 < K)
        __builtin_prefetch(&X[(mbase + row) * K + k0 + 32 + c4 * 4], 0, 1);
    }
#pragma unroll
    for (int i = 0; i < 8; ++i) {
      int linear = tid + i * 256;              // 0..2047
      int kk = linear >> 6;
      int nn = linear & 63;
      Wn[nn * XS + kk] = (_Float16)W[(k0 + kk) * N + nbase + nn];
    }
    __syncthreads();

    Frag16 a[2], b[2];
#pragma unroll
    for (int mi = 0; mi < 2; ++mi) {
      int row = wm * 32 + mi * 16 + lm;        // A: m = lane%16
      a[mi].q[0] = *(const u32x4*)&Xs[row * XS + 8 * half];
      a[mi].q[1] = *(const u32x4*)&Xs[row * XS + 16 + 8 * half];
    }
#pragma unroll
    for (int ni = 0; ni < 2; ++ni) {
      int nr = wn * 32 + ni * 16 + lm;         // B: n = lane%16
      b[ni].q[0] = *(const u32x4*)&Wn[nr * XS + 8 * half];
      b[ni].q[1] = *(const u32x4*)&Wn[nr * XS + 16 + 8 * half];
    }
#pragma unroll
    for (int mi = 0; mi < 2; ++mi)
#pragma unroll
      for (int ni = 0; ni < 2; ++ni)
        acc[mi][ni] = wmma16x16x32(a[mi].h, b[ni].h, acc[mi][ni]);
    __syncthreads();
  }

#pragma unroll
  for (int mi = 0; mi < 2; ++mi) {
#pragma unroll
    for (int ni = 0; ni < 2; ++ni) {
      int col = nbase + wn * 32 + ni * 16 + lm;
      float bb = bias[col];
#pragma unroll
      for (int r = 0; r < 8; ++r) {
        int row = mbase + wm * 32 + mi * 16 + r + 8 * half;
        Y[row * N + col] = (YT)(acc[mi][ni][r] + bb);
      }
    }
  }
}

// ---------------------------------------------------------------------------
// Flash attention: one WG (4 waves / 128 threads) handles 64 query rows of
// one (b,h); each wave owns 16 queries x d=64. Loop over 32-key blocks:
//   S = q@k^T * 1/8 (4 WMMAs) -> online softmax -> z += P@v (4 WMMAs).
// Q/K tiles staged by the Tensor Data Mover (TDM pads rows 64->72 f16);
// V tile transposed by vector loads in parallel with the in-flight TDM op.
// ---------------------------------------------------------------------------
__global__ __launch_bounds__(128)
void attn_kernel(const _Float16* __restrict__ Qp, const _Float16* __restrict__ Kp,
                 const _Float16* __restrict__ Vp, _Float16* __restrict__ Zp) {
  constexpr int S  = 2048;
  constexpr int DM = 512;
  constexpr int DH = 64;
  constexpr int QS = 72;                       // Q/K LDS row stride (f16)
  constexpr int VS = 40;                       // V/P LDS row stride (f16)

  __shared__ _Float16 smem[64 * QS + 32 * QS + 64 * VS + 4 * 16 * VS];
  _Float16* Qs = smem;                         // [query][d]
  _Float16* Ks = smem + 64 * QS;               // [key][d]   (B of q@k^T)
  _Float16* Vs = Ks + 32 * QS;                 // [d][key]   (B of p@v)
  _Float16* Ps = Vs + 64 * VS;                 // per-wave P [query][key]

  const int tid  = threadIdx.x;
  const int lane = tid & 31, wave = tid >> 5;
  const int lm   = lane & 15, half = lane >> 4;
  const int b = blockIdx.x >> 3, h = blockIdx.x & 7;
  const int qbase = blockIdx.y * 64;

  const _Float16* Qg = Qp + (size_t)(b * S + qbase) * DM + h * DH;
  const _Float16* Kg = Kp + (size_t)(b * S) * DM + h * DH;
  const _Float16* Vg = Vp + (size_t)(b * S) * DM + h * DH;

  // ---- stage Q tile 64x64 once ----
#if USE_TDM
  if (wave == 0) {
    tdm_load_tile_f16(lds_off_of(Qs), Qg, DH, 64, DM);
    __builtin_amdgcn_s_wait_tensorcnt(0);
  }
#else
#pragma unroll
  for (int i = 0; i < 8; ++i) {
    int linear = tid + i * 128;                // u32x2 units (4 f16)
    int row = linear >> 4, c = linear & 15;
    *(u32x2*)&Qs[row * QS + c * 4] = *(const u32x2*)&Qg[row * DM + c * 4];
  }
#endif
  __syncthreads();

  Frag16 aq[2];                                // A frags, d-steps 0..31 / 32..63
#pragma unroll
  for (int ks = 0; ks < 2; ++ks) {
    int row = wave * 16 + lm;
    aq[ks].q[0] = *(const u32x4*)&Qs[row * QS + ks * 32 + 8 * half];
    aq[ks].q[1] = *(const u32x4*)&Qs[row * QS + ks * 32 + 16 + 8 * half];
  }

  float m_run[8], l_run[8];
#pragma unroll
  for (int r = 0; r < 8; ++r) { m_run[r] = -1e30f; l_run[r] = 0.f; }
  v8f zacc[4] = {};                            // z[16 x 64] over 4 n-frags

  for (int kb = 0; kb < S / 32; ++kb) {
    int kbase = kb * 32;
    // ---- K tile [32 keys][64 d] via TDM (async, padded rows) ----
#if USE_TDM
    if (wave == 0)
      tdm_load_tile_f16(lds_off_of(Ks), Kg + (size_t)kbase * DM, DH, 32, DM);
#else
#pragma unroll
    for (int i = 0; i < 4; ++i) {
      int linear = tid + i * 128;              // 0..511
      int key = linear >> 4, c = linear & 15;
      *(u32x2*)&Ks[key * QS + c * 4] =
          *(const u32x2*)&Kg[(kbase + key) * DM + c * 4];
    }
#endif
    // ---- V tile transposed -> Vs[d][key] (overlaps with TDM) ----
#pragma unroll
    for (int i = 0; i < 4; ++i) {
      int linear = tid + i * 128;
      int key = linear >> 4, c = linear & 15;
      union { u32x2 u; _Float16 e[4]; } t;
      t.u = *(const u32x2*)&Vg[(kbase + key) * DM + c * 4];
#pragma unroll
      for (int j = 0; j < 4; ++j) Vs[(c * 4 + j) * VS + key] = t.e[j];
    }
#if USE_TDM
    if (wave == 0) __builtin_amdgcn_s_wait_tensorcnt(0);
#endif
    __syncthreads();

    // ---- scores: 16 queries x 32 keys, contract over d=64 ----
    v8f sc[2] = {};
#pragma unroll
    for (int ks = 0; ks < 2; ++ks) {
#pragma unroll
      for (int j = 0; j < 2; ++j) {
        Frag16 bk;
        int kr = j * 16 + lm;                  // n = key index
        bk.q[0] = *(const u32x4*)&Ks[kr * QS + ks * 32 + 8 * half];
        bk.q[1] = *(const u32x4*)&Ks[kr * QS + ks * 32 + 16 + 8 * half];
        sc[j] = wmma16x16x32(aq[ks].h, bk.h, sc[j]);
      }
    }

    // ---- online softmax (half-group ds_swizzle reductions), write P ----
    _Float16* Pw = &Ps[wave * 16 * VS];
#pragma unroll
    for (int r = 0; r < 8; ++r) {
      float v0 = sc[0][r] * 0.125f;            // 1/sqrt(64)
      float v1 = sc[1][r] * 0.125f;
      float mx = red_max16(fmaxf(v0, v1));
      float mnew = fmaxf(m_run[r], mx);
      float p0 = exp2f((v0 - mnew) * 1.44269504f);
      float p1 = exp2f((v1 - mnew) * 1.44269504f);
      float ls = red_sum16(p0 + p1);
      float f = exp2f((m_run[r] - mnew) * 1.44269504f);
      l_run[r] = l_run[r] * f + ls;
      m_run[r] = mnew;
#pragma unroll
      for (int j = 0; j < 4; ++j) zacc[j][r] *= f;
      int m = r + 8 * half;                    // C-layout row
      Pw[m * VS + lm]      = (_Float16)p0;
      Pw[m * VS + lm + 16] = (_Float16)p1;
    }

    // ---- z += P(16x32) @ V(32x64); per-wave LDS, DS in-order per wave ----
    Frag16 ap;
    ap.q[0] = *(const u32x4*)&Pw[lm * VS + 8 * half];
    ap.q[1] = *(const u32x4*)&Pw[lm * VS + 16 + 8 * half];
#pragma unroll
    for (int j = 0; j < 4; ++j) {
      Frag16 bv;
      int dr = j * 16 + lm;                    // n = d index
      bv.q[0] = *(const u32x4*)&Vs[dr * VS + 8 * half];
      bv.q[1] = *(const u32x4*)&Vs[dr * VS + 16 + 8 * half];
      zacc[j] = wmma16x16x32(ap.h, bv.h, zacc[j]);
    }
    __syncthreads();                           // protect Ks/Vs for next block
  }

  // ---- normalize and store merged-head f16 output [B,S,H*64] ----
#pragma unroll
  for (int r = 0; r < 8; ++r) {
    float inv = 1.0f / l_run[r];
    int q = qbase + wave * 16 + r + 8 * half;
    _Float16* dst = Zp + (size_t)(b * S + q) * DM + h * DH;
#pragma unroll
    for (int j = 0; j < 4; ++j)
      dst[j * 16 + lm] = (_Float16)(zacc[j][r] * inv);
  }
}

// ---------------------------------------------------------------------------
extern "C" void kernel_launch(void* const* d_in, const int* in_sizes, int n_in,
                              void* d_out, int out_size, void* d_ws, size_t ws_size,
                              hipStream_t stream) {
  const float* Q  = (const float*)d_in[0];
  const float* Kt = (const float*)d_in[1];
  const float* V  = (const float*)d_in[2];
  const float* Wq = (const float*)d_in[3];
  const float* bq = (const float*)d_in[4];
  const float* Wk = (const float*)d_in[5];
  const float* bk = (const float*)d_in[6];
  const float* Wv = (const float*)d_in[7];
  const float* bv = (const float*)d_in[8];
  const float* Wo = (const float*)d_in[9];
  const float* bo = (const float*)d_in[10];
  float* out = (float*)d_out;

  const int M  = 4 * 2048;                     // B*S
  const int N  = 512;
  const int Kd = 512;
  const size_t tsz = (size_t)M * N;            // f16 elements per tensor

  _Float16* qp = (_Float16*)d_ws;              // 4 x 8 MB f16 intermediates
  _Float16* kp = qp + tsz;
  _Float16* vp = kp + tsz;
  _Float16* zp = vp + tsz;

  dim3 gg(N / 64, M / 128), gb(256);
  gemm_bias_kernel<float, _Float16><<<gg, gb, 0, stream>>>(Q,  Wq, bq, qp, M, N, Kd);
  gemm_bias_kernel<float, _Float16><<<gg, gb, 0, stream>>>(Kt, Wk, bk, kp, M, N, Kd);
  gemm_bias_kernel<float, _Float16><<<gg, gb, 0, stream>>>(V,  Wv, bv, vp, M, N, Kd);
  attn_kernel<<<dim3(4 * 8, 2048 / 64), dim3(128), 0, stream>>>(qp, kp, vp, zp);
  gemm_bias_kernel<_Float16, float><<<gg, gb, 0, stream>>>(zp, Wo, bo, out, M, N, Kd);
}
